// FlashAttentionSlidingWindow_53068615909642
// MI455X (gfx1250) — matrix operand order repaired and checked by reading, not compiled, not run on hardware
//
#include <hip/hip_runtime.h>

// CDNA5 / gfx1250 flash-attention, causal sliding window (W=512), GQA 32q/8kv, D=128.
// Per 32-key chunk:  S^T = K x Q^T  (wmma f16->f32, queries pinned to lanes),
// online softmax (in-lane + one ds_swizzle SWAPX16), O += P x V (wmma, coalesced out).
// K/V chunks move global->LDS with the Tensor Data Mover (tensor_load_to_lds),
// double-buffered f32 staging, tracked with TENSORcnt. Boundary-only masking.

typedef __attribute__((ext_vector_type(16))) _Float16 v16h;
typedef __attribute__((ext_vector_type(8)))  float    v8f;
typedef __attribute__((ext_vector_type(4)))  unsigned int v4u;
typedef __attribute__((ext_vector_type(4)))  int      v4i;
typedef __attribute__((ext_vector_type(8)))  int      v8i;

#define S_LEN 8192
#define NHEAD 32
#define NKV   8
#define GQA   4
#define DHEAD 128
#define WIN   512
#define BR    128   // queries per workgroup (8 waves x 16)
#define BC    32    // keys per chunk

// ---- LDS layout (dynamic shared, byte offsets; all 16B aligned) ----------
#define KST_STRIDE 132                       // f32 staging row stride (128 + 4 pad dwords)
#define KST_BYTES  (BC * KST_STRIDE * 4)     // 16896 per buffer
#define OFF_KST    0                         // [2] K f32 staging
#define OFF_VST    (2 * KST_BYTES)           // [2] V f32 staging  (33792)
#define F16_STRIDE 136                       // f16 working row stride (halves)
#define OFF_K16    (OFF_VST + 2 * KST_BYTES)            // 67584
#define OFF_V16    (OFF_K16 + BC * F16_STRIDE * 2)      // 76288
#define PSTR       40                        // P per-wave row stride (halves)
#define OFF_P      (OFF_V16 + BC * F16_STRIDE * 2)      // 84992
#define SMEM_BYTES (OFF_P + 8 * 16 * PSTR * 2)          // 95232

union Frag   { uint4 q[2]; v16h h; };
union F16x16 { _Float16 e[16]; uint4 q[2]; };
union H8     { _Float16 e[8];  uint4 q; };

__device__ __forceinline__ float swz_swap16(float x) {
  // group-of-32 swizzle, xor_mask=0x10, and_mask=0x1f -> swap 16-lane halves
  return __int_as_float(__builtin_amdgcn_ds_swizzle(__float_as_int(x), 0x401F));
}
__device__ __forceinline__ float lane_bcast(float x, int srclane) {
  return __int_as_float(__builtin_amdgcn_ds_bpermute(srclane * 4, __float_as_int(x)));
}

// One TDM descriptor: 2D tile of 32 rows x 128 f32, row stride NKV*DHEAD elements,
// LDS rows padded +4 dwords (pad_interval code 6 = 128 dwords, pad_amount code 3 = 4).
__device__ __forceinline__ void tdm_issue_kv(const float* gbase, unsigned lds_off) {
  unsigned long long ga = (unsigned long long)(uintptr_t)gbase;
  v4u g0;
  g0[0] = 1u;                                     // count=1, user descriptor
  g0[1] = lds_off;                                // LDS byte address
  g0[2] = (unsigned)ga;                           // global addr [95:64]
  g0[3] = ((unsigned)(ga >> 32) & 0x1FFFFFFu) | (2u << 30);  // addr hi | type=2
  v8i g1;
  g1[0] = (int)((2u << 16) | (1u << 20) | (6u << 22) | (3u << 25)); // 4B data, pad on
  g1[1] = (int)(128u << 16);                      // tensor_dim0 = 128
  g1[2] = 0;                                      // dim0 hi | tensor_dim1 lo (65536&0xffff=0)
  g1[3] = (int)(1u | (128u << 16));               // tensor_dim1 hi=1 | tile_dim0 = 128
  g1[4] = 32;                                     // tile_dim1 = 32, tile_dim2 = 0
  g1[5] = NKV * DHEAD;                            // tensor_dim0_stride = 1024 elements
  g1[6] = 0;
  g1[7] = 0;
  v4i gz4 = (v4i)0;                               // groups 2/3 unused (2D tile)
  v8i gz8 = (v8i)0;                               // extra group (clang-23 6-arg form)
  __builtin_amdgcn_tensor_load_to_lds(g0, g1, gz4, gz4, gz8, 0);
}

__global__ __launch_bounds__(256, 1)
void swa_fwd_kernel(const float* __restrict__ qg, const float* __restrict__ kg,
                    const float* __restrict__ vg, float* __restrict__ og) {
  extern __shared__ char smem[];
  _Float16* K16 = (_Float16*)(smem + OFF_K16);
  _Float16* V16 = (_Float16*)(smem + OFF_V16);

  const int tid  = threadIdx.x;
  const int wq   = tid >> 5;        // wave id: owns 16 queries
  const int lane = tid & 31;
  const int nlo  = lane & 15;
  const int hi8  = (lane >> 4) * 8; // C-layout M offset for high lanes
  const int q0   = blockIdx.x * BR;
  const int h    = blockIdx.y;
  const int hk   = h / GQA;
  const int qwm  = q0 + wq * 16;    // wave's first query
  const int iq   = qwm + nlo;       // this lane's query in S^T layout

  // ---- Q^T B-fragments: lane%16 = query, elements = 16 contraction d (table order).
  Frag bq[4];
  {
    const float* qbase = qg + ((size_t)qwm * NHEAD + h) * DHEAD;
    #pragma unroll
    for (int ds = 0; ds < 4; ++ds) {
      F16x16 t;
      #pragma unroll
      for (int n = 0; n < 16; ++n)
        t.e[n] = (_Float16)qbase[(size_t)nlo * NHEAD * DHEAD + ds * 32 + hi8 + (n & 7) + (n >> 3) * 16];
      bq[ds].q[0] = t.q[0];
      bq[ds].q[1] = t.q[1];
    }
  }

  v8f ov[8];                        // O accumulators: M=queries, N=d (8 d-tiles)
  #pragma unroll
  for (int t = 0; t < 8; ++t) ov[t] = {};

  float m_run = -3.0e38f;
  float l_run = 0.0f;
  const float kscale = 0.08838834764831845f * 1.4426950408889634f; // 1/sqrt(128)*log2e

  const int kbeg = (q0 - WIN) < 0 ? 0 : (q0 - WIN);   // 32-aligned
  const int nch  = (q0 + BR - kbeg) / BC;             // >= 4

  // cooperative convert indices: 256 threads cover 32 rows x 128 d
  const int lr = tid >> 3;
  const int lc = (tid & 7) * 16;

  // TDM prologue: chunk 0 -> staging buffer 0 (wave 0 only; EXEC-independent op,
  // guarded by a wave-uniform scalar branch)
  if (wq == 0) {
    tdm_issue_kv(kg + ((size_t)kbeg * NKV + hk) * DHEAD, OFF_KST);
    tdm_issue_kv(vg + ((size_t)kbeg * NKV + hk) * DHEAD, OFF_VST);
  }

  for (int ch = 0; ch < nch; ++ch) {
    const int k0  = kbeg + ch * BC;
    const int buf = ch & 1;

    if (wq == 0) {
      if (ch + 1 < nch) {
        const int nb = (ch + 1) & 1;
        const size_t grow = ((size_t)(k0 + BC) * NKV + hk) * DHEAD;
        tdm_issue_kv(kg + grow, OFF_KST + nb * KST_BYTES);
        tdm_issue_kv(vg + grow, OFF_VST + nb * KST_BYTES);
        __builtin_amdgcn_s_wait_tensorcnt(2);   // in-order: chunk ch's 2 ops done
      } else {
        __builtin_amdgcn_s_wait_tensorcnt(0);
      }
    }
    __syncthreads();   // chunk ch staging visible; prior compute reads of K16/V16 done

    // -------- cooperative f32 -> f16 convert (layout preserving) ----------
    {
      const float* KstB = (const float*)(smem + OFF_KST + buf * KST_BYTES);
      const float* VstB = (const float*)(smem + OFF_VST + buf * KST_BYTES);
      const float4* ks = (const float4*)(KstB + lr * KST_STRIDE + lc);
      const float4* vs = (const float4*)(VstB + lr * KST_STRIDE + lc);
      F16x16 tk, tv;
      #pragma unroll
      for (int i = 0; i < 4; ++i) {
        float4 a = ks[i], b = vs[i];
        tk.e[4*i+0] = (_Float16)a.x; tk.e[4*i+1] = (_Float16)a.y;
        tk.e[4*i+2] = (_Float16)a.z; tk.e[4*i+3] = (_Float16)a.w;
        tv.e[4*i+0] = (_Float16)b.x; tv.e[4*i+1] = (_Float16)b.y;
        tv.e[4*i+2] = (_Float16)b.z; tv.e[4*i+3] = (_Float16)b.w;
      }
      uint4* dk = (uint4*)&K16[lr * F16_STRIDE + lc];
      uint4* dv = (uint4*)&V16[lr * F16_STRIDE + lc];
      dk[0] = tk.q[0]; dk[1] = tk.q[1];
      dv[0] = tv.q[0]; dv[1] = tv.q[1];
    }
    __syncthreads();

    // Wave-uniform band classification for this wave's 16 queries:
    //   valid keys for the wave: [qwm - (WIN-1), qwm + 15]
    const bool any_valid = (k0 <= qwm + 15) && (k0 + BC - 1 >= qwm - (WIN - 1));
    if (any_valid) {
      const bool full = (k0 + BC - 1 <= qwm) && (k0 >= qwm + 15 - (WIN - 1));

      // ---- S^T = K x Q^T : stage all 8 A-frags, then 8 back-to-back WMMA --
      Frag af[8];
      #pragma unroll
      for (int ds = 0; ds < 4; ++ds) {
        const int db = ds * 32 + hi8;
        af[2*ds+0].q[0] = *(const uint4*)&K16[nlo * F16_STRIDE + db];
        af[2*ds+0].q[1] = *(const uint4*)&K16[nlo * F16_STRIDE + db + 16];
        af[2*ds+1].q[0] = *(const uint4*)&K16[(nlo + 16) * F16_STRIDE + db];
        af[2*ds+1].q[1] = *(const uint4*)&K16[(nlo + 16) * F16_STRIDE + db + 16];
      }
      v8f c0 = {}, c1 = {};
      #pragma unroll
      for (int ds = 0; ds < 4; ++ds) {
        c0 = __builtin_amdgcn_wmma_f32_16x16x32_f16(false, af[2*ds+0].h, false, bq[ds].h,
                                                    (short)0, c0, false, false);
        c1 = __builtin_amdgcn_wmma_f32_16x16x32_f16(false, af[2*ds+1].h, false, bq[ds].h,
                                                    (short)0, c1, false, false);
      }

      // ---- mask (boundary chunks only) + online softmax -------------------
      float xs[16];
      if (full) {
        #pragma unroll
        for (int r = 0; r < 8; ++r) {
          xs[r]     = c0[r] * kscale;
          xs[r + 8] = c1[r] * kscale;
        }
      } else {
        const int jmin = iq - (WIN - 1);
        #pragma unroll
        for (int r = 0; r < 8; ++r) {
          const int j0 = k0 + hi8 + r;
          const int j1 = j0 + 16;
          xs[r]     = (j0 > iq || j0 < jmin) ? -3.0e38f : c0[r] * kscale;
          xs[r + 8] = (j1 > iq || j1 < jmin) ? -3.0e38f : c1[r] * kscale;
        }
      }
      float mx = xs[0];
      #pragma unroll
      for (int t = 1; t < 16; ++t) mx = fmaxf(mx, xs[t]);
      mx = fmaxf(mx, swz_swap16(mx));
      const float m_new = fmaxf(m_run, mx);
      const float alpha = __builtin_amdgcn_exp2f(m_run - m_new);
      float sum = 0.0f;
      #pragma unroll
      for (int t = 0; t < 16; ++t) {
        xs[t] = __builtin_amdgcn_exp2f(xs[t] - m_new);
        sum += xs[t];
      }
      sum += swz_swap16(sum);
      l_run = l_run * alpha + sum;
      m_run = m_new;

      // redistribute alpha to O-layout rows (query m = r + hi8) and rescale O
      float ar[8];
      #pragma unroll
      for (int r = 0; r < 8; ++r) ar[r] = lane_bcast(alpha, r + hi8);
      #pragma unroll
      for (int dt = 0; dt < 8; ++dt) {
        #pragma unroll
        for (int r = 0; r < 8; ++r) ov[dt][r] *= ar[r];
      }

      // ---- stage P row-major [query][key] (two packed b128 per lane) ------
      _Float16* ptw = (_Float16*)(smem + OFF_P) + wq * 16 * PSTR;
      H8 p0, p1;
      #pragma unroll
      for (int r = 0; r < 8; ++r) { p0.e[r] = (_Float16)xs[r]; p1.e[r] = (_Float16)xs[r + 8]; }
      *(uint4*)((char*)ptw + nlo * (PSTR * 2) + hi8 * 2)      = p0.q;  // keys hi8..hi8+7
      *(uint4*)((char*)ptw + nlo * (PSTR * 2) + 32 + hi8 * 2) = p1.q;  // keys 16+hi8..

      Frag ap;   // A: M = 16 queries, K = 32 keys
      ap.q[0] = *(const uint4*)((const char*)ptw + nlo * (PSTR * 2) + (lane >> 4) * 16);
      ap.q[1] = *(const uint4*)((const char*)ptw + nlo * (PSTR * 2) + 32 + (lane >> 4) * 16);

      // ---- O += P x V : stage all 8 B-frags, then 8 back-to-back WMMA -----
      Frag bf[8];
      #pragma unroll
      for (int dt = 0; dt < 8; ++dt) {
        #pragma unroll
        for (int s = 0; s < 16; ++s)
          bf[dt].h[s] = V16[(hi8 + (s & 7) + (s >> 3) * 16) * F16_STRIDE + dt * 16 + nlo];
      }
      #pragma unroll
      for (int dt = 0; dt < 8; ++dt) {
        ov[dt] = __builtin_amdgcn_wmma_f32_16x16x32_f16(false, ap.h, false, bf[dt].h,
                                                        (short)0, ov[dt], false, false);
      }
    }
  }

  // -------- normalize and store O (coalesced: lanes = consecutive d) ------
  const float invl = 1.0f / l_run;
  float ir[8];
  #pragma unroll
  for (int r = 0; r < 8; ++r) ir[r] = lane_bcast(invl, r + hi8);
  float* ob = og + ((size_t)(qwm + hi8) * NHEAD + h) * DHEAD + nlo;
  #pragma unroll
  for (int r = 0; r < 8; ++r) {
    float* obr = ob + (size_t)r * NHEAD * DHEAD;
    #pragma unroll
    for (int dt = 0; dt < 8; ++dt) obr[dt * 16] = ov[dt][r] * ir[r];
  }
}

extern "C" void kernel_launch(void* const* d_in, const int* in_sizes, int n_in,
                              void* d_out, int out_size, void* d_ws, size_t ws_size,
                              hipStream_t stream) {
  const float* q = (const float*)d_in[0];
  const float* k = (const float*)d_in[1];
  const float* v = (const float*)d_in[2];
  float* o = (float*)d_out;
  (void)in_sizes; (void)n_in; (void)out_size; (void)d_ws; (void)ws_size;
  (void)hipFuncSetAttribute((const void*)swa_fwd_kernel,
                            hipFuncAttributeMaxDynamicSharedMemorySize, SMEM_BYTES);
  dim3 grid(S_LEN / BR, NHEAD);
  swa_fwd_kernel<<<grid, dim3(256), SMEM_BYTES, stream>>>(q, k, v, o);
}
